// RecurrentRouter_82231443849807
// MI455X (gfx1250) — compile-verified, matrix operand order checked by reading, not run on hardware
//
#include <hip/hip_runtime.h>
#include <hip/hip_bf16.h>
#include <math.h>
#include <stdint.h>

// MI455X / gfx1250, wave32. WMMA bf16 16x16x32 f32-acc; async global->LDS staging.

typedef __attribute__((ext_vector_type(16))) __bf16 v16bf;
typedef __attribute__((ext_vector_type(8)))  __bf16 v8bf;
typedef __attribute__((ext_vector_type(8)))  float  v8f;

#define B_  8
#define S_  2048
#define D_  2048
#define H_  512
#define E_  64
#define M_  (B_*S_)   // 16384 tokens

// ---------------------------------------------------------------------------
// f32 -> bf16 conversion (weight packing)
// ---------------------------------------------------------------------------
__global__ __launch_bounds__(256)
void f32_to_bf16_kernel(const float* __restrict__ src, __bf16* __restrict__ dst, int n) {
    int i = blockIdx.x * blockDim.x + threadIdx.x;
    if (i < n) dst[i] = (__bf16)src[i];
}

// ---------------------------------------------------------------------------
// LayerNorm over D=2048, emit bf16 activations for the WMMA GEMM
// ---------------------------------------------------------------------------
__global__ __launch_bounds__(256)
void layernorm_bf16_kernel(const float* __restrict__ x, const float* __restrict__ g,
                           const float* __restrict__ bta, __bf16* __restrict__ out) {
    const int row = blockIdx.x;
    const int tid = threadIdx.x;
    const float* xr = x + (size_t)row * D_;
    float xv[8];
    float s = 0.f, sq = 0.f;
#pragma unroll
    for (int i = 0; i < 8; ++i) {
        float v = xr[tid + i * 256];
        xv[i] = v; s += v; sq += v * v;
    }
    __shared__ float r1[256], r2[256];
    r1[tid] = s; r2[tid] = sq; __syncthreads();
    for (int st = 128; st > 0; st >>= 1) {
        if (tid < st) { r1[tid] += r1[tid + st]; r2[tid] += r2[tid + st]; }
        __syncthreads();
    }
    const float mu   = r1[0] * (1.0f / D_);
    const float var  = r2[0] * (1.0f / D_) - mu * mu;
    const float rstd = rsqrtf(var + 1e-5f);
    __bf16* orow = out + (size_t)row * D_;
#pragma unroll
    for (int i = 0; i < 8; ++i) {
        int c = tid + i * 256;
        orow[c] = (__bf16)((xv[i] - mu) * rstd * g[c] + bta[c]);
    }
}

// ---------------------------------------------------------------------------
// WMMA GEMM: out = act(A(MxK) @ W^T(NxK) + bias)
// Block = 8 waves, tile 128(M) x 64(N). B panel is block-uniform, so it is
// staged global->LDS with gfx1250 async-DMA (global_load_async_to_lds_b128,
// ASYNCcnt) and double-buffered in K-chunks of 128. A streams per-wave from
// global (one v16bf per k-step). act: 0=none, 1=exact GELU.
// ---------------------------------------------------------------------------
#define KC   128          // K-chunk
#define KCP  (KC + 8)     // LDS row pitch (bf16 elems): 272B -> bank-spread, 16B aligned

__global__ __launch_bounds__(256)
void wmma_gemm_kernel(const __bf16* __restrict__ A, const __bf16* __restrict__ W,
                      const float* __restrict__ bias, void* __restrict__ out,
                      int M, int N, int K, int act, int out_is_bf16) {
    __shared__ __bf16 Bs[2][64 * KCP];   // 2 x 17KB

    const int tid     = threadIdx.x;
    const int lane    = tid & 31;
    const int wave    = tid >> 5;
    const int m_base  = blockIdx.x * 128 + wave * 16;
    const int n_base  = blockIdx.y * 64;
    const int halfsel = lane >> 4;   // K sub-slab 0..15 / 16..31
    const int idx16   = lane & 15;

    const __bf16* arow = A + (size_t)(m_base + idx16) * K + halfsel * 16;

    // async-DMA one 64col x 128k B chunk into LDS buffer `buf`
    auto issue_chunk = [&](int buf, int kc) {
#pragma unroll
        for (int i = 0; i < 4; ++i) {
            const int slot = i * 256 + tid;         // 1024 x 16B segments
            const int col  = slot >> 4;             // 0..63
            const int seg  = slot & 15;             // 16B segment within 256B row
            const __bf16* gp = W + (size_t)(n_base + col) * K + kc + seg * 8;
            const unsigned la =
                (unsigned)(uintptr_t)(&Bs[buf][col * KCP]) + (unsigned)(seg * 16);
            asm volatile("global_load_async_to_lds_b128 %0, %1, off"
                         :: "v"(la), "v"(gp) : "memory");
        }
    };

    v8f acc[4] = {v8f{}, v8f{}, v8f{}, v8f{}};

    issue_chunk(0, 0);
    const int NC = K / KC;
    for (int c = 0; c < NC; ++c) {
        const int cur = c & 1;
        const int kc  = c * KC;
        asm volatile("s_wait_asynccnt 0x0" ::: "memory");  // chunk c landed (this wave)
        __syncthreads();                                   // ...for all waves
        if (c + 1 < NC) issue_chunk(cur ^ 1, kc + KC);     // overlap next chunk's DMA

        // preload this chunk's A slabs (4 x v16bf)
        v16bf a_reg[KC / 32];
#pragma unroll
        for (int kk = 0; kk < KC; kk += 32) {
            a_reg[kk / 32] = *(const v16bf*)(arow + kc + kk);
            __builtin_prefetch(arow + kc + kk + KC, 0, 1);
        }

#pragma unroll
        for (int kk = 0; kk < KC; kk += 32) {
#pragma unroll
            for (int t = 0; t < 4; ++t) {
                const __bf16* bp =
                    &Bs[cur][(t * 16 + idx16) * KCP + kk + halfsel * 16];
                v8bf lo = *(const v8bf*)(bp);
                v8bf hi = *(const v8bf*)(bp + 8);
                v16bf b = __builtin_shufflevector(lo, hi, 0, 1, 2, 3, 4, 5, 6, 7,
                                                  8, 9, 10, 11, 12, 13, 14, 15);
                acc[t] = __builtin_amdgcn_wmma_f32_16x16x32_bf16(
                    false, a_reg[kk / 32], false, b, (short)0, acc[t], false, false);
            }
        }
        __syncthreads();   // done reading Bs[cur] before it is DMA'd over next pass
    }

#pragma unroll
    for (int t = 0; t < 4; ++t) {
        const int col = n_base + t * 16 + idx16;
        const float bb = bias ? bias[col] : 0.0f;
#pragma unroll
        for (int r = 0; r < 8; ++r) {
            const int m = m_base + r + halfsel * 8;  // C layout: VGPR r = rows r, r+8
            float v = acc[t][r] + bb;
            if (act == 1) v = 0.5f * v * (1.0f + erff(v * 0.70710678118654752f));
            if (out_is_bf16) ((__bf16*)out)[(size_t)m * N + col] = (__bf16)v;
            else             ((float*)out)[(size_t)m * N + col] = v;
        }
    }
}

// ---------------------------------------------------------------------------
// GRU scan: 2048 serial steps, single workgroup (32 waves), h resident in LDS
// (f32 master + bf16 WMMA copy). Batch 8 padded to one WMMA M-tile of 16.
// Each wave owns 16 hidden columns + all 3 gates; Wh (1.5MB bf16) streams from
// L2 each step. K loop is register double-buffered to overlap loads with WMMA.
// ---------------------------------------------------------------------------
__global__ __launch_bounds__(1024)
void gru_scan_kernel(const float* __restrict__ gi,      // (B,S,3H) f32, gates r,z,n
                     const __bf16* __restrict__ Wh,     // (3H,H) bf16 packed r,z,n
                     const float* __restrict__ b_hr,
                     const float* __restrict__ b_hz,
                     const float* __restrict__ b_hn,
                     __bf16* __restrict__ hs,           // (B,S,H) bf16
                     float* __restrict__ h_final) {     // (B,H) f32
    __shared__ __bf16 h_bf[16 * H_];   // 16 KB
    __shared__ float  h_f [16 * H_];   // 32 KB
    const int tid = threadIdx.x;
    for (int i = tid; i < 16 * H_; i += 1024) { h_bf[i] = (__bf16)0.0f; h_f[i] = 0.0f; }
    __syncthreads();

    const int lane    = tid & 31;
    const int wave    = tid >> 5;        // 0..31
    const int halfsel = lane >> 4;
    const int idx16   = lane & 15;
    const int col     = wave * 16 + idx16;   // hidden column this lane covers

    const __bf16* wr = Wh + (size_t)(0 * H_ + col) * H_ + halfsel * 16;
    const __bf16* wz = Wh + (size_t)(1 * H_ + col) * H_ + halfsel * 16;
    const __bf16* wn = Wh + (size_t)(2 * H_ + col) * H_ + halfsel * 16;
    const __bf16* ha = &h_bf[idx16 * H_ + halfsel * 16];

    const float bhr = b_hr[col], bhz = b_hz[col], bhn = b_hn[col];

    for (int t = 0; t < S_; ++t) {
        v8f ar = {}, az = {}, an = {};
        // software-pipelined K loop: load k+32 while WMMA-ing k
        v16bf a0 = *(const v16bf*)(ha);
        v16bf r0 = *(const v16bf*)(wr);
        v16bf z0 = *(const v16bf*)(wz);
        v16bf n0 = *(const v16bf*)(wn);
#pragma unroll 4
        for (int k = 0; k < H_; k += 32) {
            v16bf a1 = a0, r1 = r0, z1 = z0, n1 = n0;
            if (k + 32 < H_) {
                a1 = *(const v16bf*)(ha + k + 32);
                r1 = *(const v16bf*)(wr + k + 32);
                z1 = *(const v16bf*)(wz + k + 32);
                n1 = *(const v16bf*)(wn + k + 32);
            }
            ar = __builtin_amdgcn_wmma_f32_16x16x32_bf16(false, a0, false, r0, (short)0, ar, false, false);
            az = __builtin_amdgcn_wmma_f32_16x16x32_bf16(false, a0, false, z0, (short)0, az, false, false);
            an = __builtin_amdgcn_wmma_f32_16x16x32_bf16(false, a0, false, n0, (short)0, an, false, false);
            a0 = a1; r0 = r1; z0 = z1; n0 = n1;
        }
        __syncthreads();   // everyone done reading h_bf for this step

        if (halfsel == 0) {
            // real batch rows 0..7 live in acc elements 0..7 on lanes 0..15
#pragma unroll
            for (int r = 0; r < 8; ++r) {
                const size_t gbase = ((size_t)r * S_ + t) * (3 * H_) + col;
                const float gr = gi[gbase         ] + ar[r] + bhr;
                const float gz = gi[gbase +     H_] + az[r] + bhz;
                const float gn = gi[gbase + 2 * H_];
                const float rv = 1.0f / (1.0f + expf(-gr));
                const float zv = 1.0f / (1.0f + expf(-gz));
                const float nv = tanhf(gn + rv * (an[r] + bhn));
                const float hold = h_f[r * H_ + col];
                const float hnew = (1.0f - zv) * nv + zv * hold;
                h_f[r * H_ + col] = hnew;
                const __bf16 hb = (__bf16)hnew;
                h_bf[r * H_ + col] = hb;
                hs[((size_t)r * S_ + t) * H_ + col] = hb;
            }
        }
        __syncthreads();   // h updated before next step's WMMA reads
    }

    for (int i = tid; i < B_ * H_; i += 1024) h_final[i] = h_f[i];
}

// ---------------------------------------------------------------------------
// Temperature softmax over E=64
// ---------------------------------------------------------------------------
__global__ __launch_bounds__(64)
void softmax_kernel(const float* __restrict__ logits, const float* __restrict__ temp,
                    float* __restrict__ probs) {
    const int row = blockIdx.x;
    const int tid = threadIdx.x;
    const float inv = 1.0f / (fabsf(temp[0]) + 1e-6f);
    const float v = logits[(size_t)row * E_ + tid] * inv;
    __shared__ float red[64];
    red[tid] = v; __syncthreads();
    for (int s = 32; s > 0; s >>= 1) {
        if (tid < s) red[tid] = fmaxf(red[tid], red[tid + s]);
        __syncthreads();
    }
    const float mx = red[0]; __syncthreads();
    const float e = expf(v - mx);
    red[tid] = e; __syncthreads();
    for (int s = 32; s > 0; s >>= 1) {
        if (tid < s) red[tid] += red[tid + s];
        __syncthreads();
    }
    probs[(size_t)row * E_ + tid] = e / red[0];
}

// ---------------------------------------------------------------------------
// Host-side orchestration
// ---------------------------------------------------------------------------
extern "C" void kernel_launch(void* const* d_in, const int* in_sizes, int n_in,
                              void* d_out, int out_size, void* d_ws, size_t ws_size,
                              hipStream_t stream) {
    const float* x     = (const float*)d_in[0];
    const float* ln_g  = (const float*)d_in[1];
    const float* ln_b  = (const float*)d_in[2];
    const float* W_p   = (const float*)d_in[3];
    const float* b_p   = (const float*)d_in[4];
    const float* W_ir  = (const float*)d_in[5];
    const float* W_hr  = (const float*)d_in[6];
    const float* b_ir  = (const float*)d_in[7];
    const float* b_hr  = (const float*)d_in[8];
    const float* W_iz  = (const float*)d_in[9];
    const float* W_hz  = (const float*)d_in[10];
    const float* b_iz  = (const float*)d_in[11];
    const float* b_hz  = (const float*)d_in[12];
    const float* W_in  = (const float*)d_in[13];
    const float* W_hn  = (const float*)d_in[14];
    const float* b_in  = (const float*)d_in[15];
    const float* b_hn  = (const float*)d_in[16];
    const float* W_out = (const float*)d_in[17];
    const float* b_out = (const float*)d_in[18];
    const float* temp  = (const float*)d_in[19];

    char* ws = (char*)d_ws;
    size_t off = 0;
    auto carve = [&](size_t bytes) -> char* {
        char* p = ws + off;
        off = (off + bytes + 255) & ~(size_t)255;
        return p;
    };

    __bf16* xn_bf = (__bf16*)carve((size_t)M_ * D_ * 2);      // 64 MB
    __bf16* Wp_bf = (__bf16*)carve((size_t)H_ * D_ * 2);      //  2 MB
    __bf16* Wi_bf = (__bf16*)carve((size_t)3 * H_ * H_ * 2);  // 1.5 MB
    __bf16* Wh_bf = (__bf16*)carve((size_t)3 * H_ * H_ * 2);  // 1.5 MB
    __bf16* Wo_bf = (__bf16*)carve((size_t)E_ * H_ * 2);      // 64 KB
    __bf16* p_bf  = (__bf16*)carve((size_t)M_ * H_ * 2);      // 16 MB
    float*  gi_f  = (float*) carve((size_t)M_ * 3 * H_ * 4);  // 96 MB
    __bf16* hs_bf = (__bf16*)carve((size_t)M_ * H_ * 2);      // 16 MB
    float*  bi_pk = (float*) carve((size_t)3 * H_ * 4);       //  6 KB

    // Weight conversions / packing (r,z,n gate order)
    {
        int n = H_ * D_;
        f32_to_bf16_kernel<<<(n + 255) / 256, 256, 0, stream>>>(W_p, Wp_bf, n);
        n = H_ * H_;
        f32_to_bf16_kernel<<<(n + 255) / 256, 256, 0, stream>>>(W_ir, Wi_bf + 0 * (size_t)H_ * H_, n);
        f32_to_bf16_kernel<<<(n + 255) / 256, 256, 0, stream>>>(W_iz, Wi_bf + 1 * (size_t)H_ * H_, n);
        f32_to_bf16_kernel<<<(n + 255) / 256, 256, 0, stream>>>(W_in, Wi_bf + 2 * (size_t)H_ * H_, n);
        f32_to_bf16_kernel<<<(n + 255) / 256, 256, 0, stream>>>(W_hr, Wh_bf + 0 * (size_t)H_ * H_, n);
        f32_to_bf16_kernel<<<(n + 255) / 256, 256, 0, stream>>>(W_hz, Wh_bf + 1 * (size_t)H_ * H_, n);
        f32_to_bf16_kernel<<<(n + 255) / 256, 256, 0, stream>>>(W_hn, Wh_bf + 2 * (size_t)H_ * H_, n);
        n = E_ * H_;
        f32_to_bf16_kernel<<<(n + 255) / 256, 256, 0, stream>>>(W_out, Wo_bf, n);
    }
    hipMemcpyAsync(bi_pk + 0 * H_, b_ir, H_ * sizeof(float), hipMemcpyDeviceToDevice, stream);
    hipMemcpyAsync(bi_pk + 1 * H_, b_iz, H_ * sizeof(float), hipMemcpyDeviceToDevice, stream);
    hipMemcpyAsync(bi_pk + 2 * H_, b_in, H_ * sizeof(float), hipMemcpyDeviceToDevice, stream);

    // 1) LayerNorm -> bf16
    layernorm_bf16_kernel<<<M_, 256, 0, stream>>>(x, ln_g, ln_b, xn_bf);

    // 2) p = GELU(xn @ Wp^T + b_p), bf16 out
    { dim3 g(M_ / 128, H_ / 64);
      wmma_gemm_kernel<<<g, 256, 0, stream>>>(xn_bf, Wp_bf, b_p, p_bf, M_, H_, D_, 1, 1); }

    // 3) gi = p @ [Wir;Wiz;Win]^T + bias (fused N=1536), f32 out
    { dim3 g(M_ / 128, (3 * H_) / 64);
      wmma_gemm_kernel<<<g, 256, 0, stream>>>(p_bf, Wi_bf, bi_pk, gi_f, M_, 3 * H_, H_, 0, 0); }

    float* out_f   = (float*)d_out;
    float* logits  = out_f;
    float* probs   = out_f + (size_t)M_ * E_;
    float* h_final = out_f + (size_t)2 * M_ * E_;

    // 4) serial GRU scan (WMMA per step, h in LDS)
    gru_scan_kernel<<<1, 1024, 0, stream>>>(gi_f, Wh_bf, b_hr, b_hz, b_hn, hs_bf, h_final);

    // 5) router logits = hs @ Wout^T + b_out
    { dim3 g(M_ / 128, E_ / 64);
      wmma_gemm_kernel<<<g, 256, 0, stream>>>(hs_bf, Wo_bf, b_out, logits, M_, E_, H_, 0, 0); }

    // 6) temperature softmax
    softmax_kernel<<<M_, 64, 0, stream>>>(logits, temp, probs);
}
